// ElementNNModel_34797825032477
// MI455X (gfx1250) — matrix operand order repaired and checked by reading, not compiled, not run on hardware
//
#include <hip/hip_runtime.h>

#define N_ATOMS 131072
#define D_IN    512
#define D_HID   512
#define N_TYPES 4

typedef __attribute__((ext_vector_type(16))) _Float16 v16h;
typedef __attribute__((ext_vector_type(8)))  _Float16 v8h;
typedef __attribute__((ext_vector_type(8)))  float    v8f;

#if __has_builtin(__builtin_amdgcn_sched_barrier)
#define SCHED_FENCE() __builtin_amdgcn_sched_barrier(0)
#else
#define SCHED_FENCE() asm volatile("" ::: "memory")
#endif

// ---------------- routing: counting sort by species ----------------

__global__ void init_counts_kernel(int* cnt) {
    if (threadIdx.x < N_TYPES) cnt[threadIdx.x] = 0;
}

__global__ void count_species_kernel(const int* __restrict__ species, int* cnt) {
    int i = blockIdx.x * blockDim.x + threadIdx.x;
    if (i < N_ATOMS) atomicAdd(&cnt[species[i]], 1);
}

__global__ void scan_counts_kernel(const int* __restrict__ cnt, int* offs, int* curs) {
    if (threadIdx.x == 0) {
        int acc = 0;
        for (int e = 0; e < N_TYPES; ++e) { offs[e] = acc; curs[e] = acc; acc += cnt[e]; }
        offs[N_TYPES] = acc;
    }
}

__global__ void scatter_perm_kernel(const int* __restrict__ species, int* curs,
                                    int* __restrict__ perm) {
    int i = blockIdx.x * blockDim.x + threadIdx.x;
    if (i < N_ATOMS) {
        int pos = atomicAdd(&curs[species[i]], 1);
        perm[pos] = i;
    }
}

// ---------------- weight pre-transpose: Wt[e][n][k] (f16) = W[e][k][n] (f32) --

__global__ void convert_w_kernel(const float* __restrict__ W, _Float16* __restrict__ Wt) {
    int n = blockIdx.x;   // 0..511
    int e = blockIdx.y;   // 0..3
    const float* src = W + (size_t)e * D_HID * D_HID + n;
    _Float16*    dst = Wt + ((size_t)e * D_HID + n) * D_HID;
    for (int k = threadIdx.x; k < D_HID; k += blockDim.x)
        dst[k] = (_Float16)src[(size_t)k * D_HID];
}

__device__ __forceinline__ float silu(float x) { return x / (1.0f + __expf(-x)); }

// ---------------- fused GEMM + bias + SiLU ----------------
// Block = 8 waves, tile M=32 x N=512 (full width -> each A row read ONCE).
// Wave (wr = w>>2, wc = w&3) owns a 16 x 128 strip: 8 WMMA C tiles.
// A sched_barrier pins the per-K-step order to "all loads+cvts, then 8 WMMAs":
// the 8 B fragments stay live in distinct VGPRs, their 16 b128 loads issue as
// one clause, and the WMMA chain overlaps the memory pipeline instead of
// fencing on s_wait_loadcnt 0 before every WMMA.

template <bool GATHER>
__global__ __launch_bounds__(256)
void gemm_silu_kernel(const void* __restrict__ Ain,          // f32 X (GATHER) / f16 H
                      const _Float16* __restrict__ Wt,       // [e][n][k] f16
                      const float* __restrict__ bias,        // [e][n] f32
                      const int* __restrict__ perm,
                      const int* __restrict__ offs,
                      _Float16* __restrict__ Hout) {
    const int e = blockIdx.z;
    const int seg_start = offs[e];
    const int seg_len   = offs[e + 1] - seg_start;
    const int mtile0    = blockIdx.x * 32;
    if (mtile0 >= seg_len) return;            // uniform exit: EXEC stays full for WMMA

    const int lane  = threadIdx.x & 31;
    const int wave  = threadIdx.x >> 5;
    const int mtile = mtile0 + (wave >> 2) * 16;   // wave M base (in segment)
    const int n0    = (wave & 3) * 128;            // wave N base (block spans full 512)
    const int col   = lane & 15;
    const int kh    = (lane >> 4) * 8;             // A: hi-half lanes shift K by 8

    // per-lane B base: column (n0+col), K offset (lane>=16 ? 8 : 0) in each 32-chunk
    const _Float16* bptr =
        Wt + ((size_t)e * D_HID + n0 + col) * D_HID + (lane >> 4) * 8;

    // A source row (clamped; out-of-segment rows discarded in epilogue)
    const int mrow   = mtile + (lane & 15);
    const int srcPos = seg_start + ((mrow < seg_len) ? mrow : 0);
    const float*    xrow = GATHER ? ((const float*)Ain + (size_t)perm[srcPos] * D_IN) : nullptr;
    const _Float16* hrow = GATHER ? nullptr : ((const _Float16*)Ain + (size_t)srcPos * D_HID);

    v8f c[8] = {};
    for (int kb = 0; kb < D_IN; kb += 32) {
        // ---- operand stage: A fragment + all 8 B fragments ----
        v16h a;
        if (GATHER) {
            const float* p = xrow + kb + kh;
            float4 a0 = *(const float4*)(p);
            float4 a1 = *(const float4*)(p + 4);
            float4 a2 = *(const float4*)(p + 16);
            float4 a3 = *(const float4*)(p + 20);
            a[0]=(_Float16)a0.x;  a[1]=(_Float16)a0.y;  a[2]=(_Float16)a0.z;  a[3]=(_Float16)a0.w;
            a[4]=(_Float16)a1.x;  a[5]=(_Float16)a1.y;  a[6]=(_Float16)a1.z;  a[7]=(_Float16)a1.w;
            a[8]=(_Float16)a2.x;  a[9]=(_Float16)a2.y;  a[10]=(_Float16)a2.z; a[11]=(_Float16)a2.w;
            a[12]=(_Float16)a3.x; a[13]=(_Float16)a3.y; a[14]=(_Float16)a3.z; a[15]=(_Float16)a3.w;
        } else {
            v8h lo = *(const v8h*)(hrow + kb + kh);
            v8h hi = *(const v8h*)(hrow + kb + kh + 16);
            #pragma unroll
            for (int i = 0; i < 8; ++i) { a[i] = lo[i]; a[8 + i] = hi[i]; }
        }
        v16h b[8];
        #pragma unroll
        for (int nb = 0; nb < 8; ++nb) {
            v8h blo = *(const v8h*)(bptr + (size_t)nb * 16 * D_HID + kb);
            v8h bhi = *(const v8h*)(bptr + (size_t)nb * 16 * D_HID + kb + 16);
            #pragma unroll
            for (int i = 0; i < 8; ++i) { b[nb][i] = blo[i]; b[nb][8 + i] = bhi[i]; }
        }

        // ---- fence: nothing may cross; keeps all B fragments live ----
        SCHED_FENCE();

        // ---- 8 back-to-back WMMAs with staggered loadcnt waits ----
        #pragma unroll
        for (int nb = 0; nb < 8; ++nb)
            c[nb] = __builtin_amdgcn_wmma_f32_16x16x32_f16(
                        false, a, false, b[nb], (short)0, c[nb], false, false);

        SCHED_FENCE();   // keep next iteration's loads after this WMMA block
    }

    // epilogue: bias + SiLU, f16 store at sorted position
    const int rh = (lane >> 4) * 8;
    #pragma unroll
    for (int nb = 0; nb < 8; ++nb) {
        const int n = n0 + nb * 16 + col;
        const float bn = bias[(size_t)e * D_HID + n];
        #pragma unroll
        for (int r = 0; r < 8; ++r) {
            int m = mtile + rh + r;
            if (m < seg_len)
                Hout[(size_t)(seg_start + m) * D_HID + n] = (_Float16)silu(c[nb][r] + bn);
        }
    }
}

// ---------------- layer 3: out[perm[p]] = H2[p] . w3[e] + b3[e] -------------

__global__ __launch_bounds__(256)
void out_layer_kernel(const _Float16* __restrict__ H2, const float* __restrict__ W3,
                      const float* __restrict__ b3, const int* __restrict__ perm,
                      const int* __restrict__ offs, float* __restrict__ out) {
    int p    = blockIdx.x * 8 + (threadIdx.x >> 5);   // one wave per sorted row
    int lane = threadIdx.x & 31;
    if (p >= N_ATOMS) return;

    int e = 0;
    #pragma unroll
    for (int t = 0; t < N_TYPES - 1; ++t) e += (p >= offs[t + 1]) ? 1 : 0;

    const _Float16* h = H2 + (size_t)p * D_HID;
    const float*    w = W3 + (size_t)e * D_HID;
    float acc = 0.0f;
    #pragma unroll 4
    for (int j = lane; j < D_HID; j += 32) acc += (float)h[j] * w[j];
    #pragma unroll
    for (int s = 16; s > 0; s >>= 1) acc += __shfl_xor(acc, s, 32);
    if (lane == 0) out[perm[p]] = acc + b3[e];
}

// ---------------- host launcher ----------------

extern "C" void kernel_launch(void* const* d_in, const int* in_sizes, int n_in,
                              void* d_out, int out_size, void* d_ws, size_t ws_size,
                              hipStream_t stream) {
    const float* density = (const float*)d_in[0];
    const int*   species = (const int*)  d_in[1];
    const float* W1      = (const float*)d_in[2];
    const float* b1      = (const float*)d_in[3];
    const float* W2      = (const float*)d_in[4];
    const float* b2      = (const float*)d_in[5];
    const float* W3      = (const float*)d_in[6];
    const float* b3      = (const float*)d_in[7];
    float* out = (float*)d_out;

    char* ws = (char*)d_ws;
    int*      cnt  = (int*)(ws + 0);
    int*      offs = (int*)(ws + 64);
    int*      curs = (int*)(ws + 128);
    int*      perm = (int*)(ws + 256);                 // 512 KB
    _Float16* W1t  = (_Float16*)(ws + (1ull  << 20));  // 2 MB
    _Float16* W2t  = (_Float16*)(ws + (3ull  << 20));  // 2 MB
    _Float16* H1   = (_Float16*)(ws + (8ull  << 20));  // 128 MB
    _Float16* H2   = (_Float16*)(ws + (136ull << 20)); // 128 MB

    init_counts_kernel  <<<1, 32, 0, stream>>>(cnt);
    count_species_kernel<<<N_ATOMS / 256, 256, 0, stream>>>(species, cnt);
    scan_counts_kernel  <<<1, 32, 0, stream>>>(cnt, offs, curs);
    scatter_perm_kernel <<<N_ATOMS / 256, 256, 0, stream>>>(species, curs, perm);

    convert_w_kernel<<<dim3(D_HID, N_TYPES), 256, 0, stream>>>(W1, W1t);
    convert_w_kernel<<<dim3(D_HID, N_TYPES), 256, 0, stream>>>(W2, W2t);

    dim3 g(N_ATOMS / 32, 1, N_TYPES);  // M-tiles cover worst-case segment; full N per block
    gemm_silu_kernel<true ><<<g, 256, 0, stream>>>(density, W1t, b1, perm, offs, H1);
    gemm_silu_kernel<false><<<g, 256, 0, stream>>>(H1,      W2t, b2, perm, offs, H2);

    out_layer_kernel<<<N_ATOMS / 8, 256, 0, stream>>>(H2, W3, b3, perm, offs, out);
}
